// ResidualSpectralSpatialBlock_37726992728100
// MI455X (gfx1250) — matrix operand order repaired and checked by reading, not compiled, not run on hardware
//
#include <hip/hip_runtime.h>
#include <hip/hip_bf16.h>

// ---------------- problem constants ----------------
#define BB      8
#define LL      256
#define DM      256
#define DS      16
#define DI      512
#define DTR     16
#define DCONV   4
#define MROWS   (BB * LL)          // 2048

typedef __attribute__((ext_vector_type(16))) __bf16 v16bf;
typedef __attribute__((ext_vector_type(8)))  float  v8f;

// ---------------- bf16 split helpers (RNE) ----------------
__device__ __forceinline__ unsigned short f2bf_bits(float f) {
    unsigned u = __builtin_bit_cast(unsigned, f);
    unsigned r = u + 0x7FFFu + ((u >> 16) & 1u);          // round-to-nearest-even
    return (unsigned short)(r >> 16);
}
__device__ __forceinline__ void split_hi_lo(float f, unsigned short* hi, unsigned short* lo) {
    unsigned short h = f2bf_bits(f);
    float hf = __builtin_bit_cast(float, ((unsigned)h) << 16);
    *hi = h;
    *lo = f2bf_bits(f - hf);
}

// fragment loaders: 16 bf16 elements as two 16-byte vector loads
struct U8x32 { uint4 a, b; };
__device__ __forceinline__ v16bf ld_frag_a(const unsigned short* p) {
    // A fragment: elements 0..15 at p[0..15] (contiguous)
    U8x32 u;
    u.a = *(const uint4*)(p);
    u.b = *(const uint4*)(p + 8);
    return __builtin_bit_cast(v16bf, u);
}
__device__ __forceinline__ v16bf ld_frag_b(const unsigned short* p) {
    // B fragment: 16 contiguous ushorts (fragment-major packed)
    U8x32 u;
    u.a = *(const uint4*)(p);
    u.b = *(const uint4*)(p + 8);
    return __builtin_bit_cast(v16bf, u);
}

// ---------------- pre-pass: split fp32 activation -> bf16 hi/lo ------------
__global__ void split_a_kernel(const float* __restrict__ x,
                               unsigned short* __restrict__ hi,
                               unsigned short* __restrict__ lo, int n) {
    int i = (int)(blockIdx.x * blockDim.x + threadIdx.x);
    if (i >= n) return;
    unsigned short h, l;
    split_hi_lo(x[i], &h, &l);
    hi[i] = h; lo[i] = l;
}

// ---------------- pre-pass: pack fp32 weight (KxN) -> fragment-major bf16 --
// dest layout: [(kt*N + c)*2 + half]*16 + j   <-  B[kt*32 + half*16 + j, c]
__global__ void pack_b_kernel(const float* __restrict__ B,
                              unsigned short* __restrict__ bh,
                              unsigned short* __restrict__ bl, int N, int K) {
    int idx = (int)(blockIdx.x * blockDim.x + threadIdx.x);
    if (idx >= K * N) return;
    int j    = idx & 15;
    int half = (idx >> 4) & 1;
    int rest = idx >> 5;
    int c    = rest % N;
    int kt   = rest / N;
    int k    = kt * 32 + half * 16 + j;
    unsigned short h, l;
    split_hi_lo(B[(size_t)k * N + c], &h, &l);
    bh[idx] = h; bl[idx] = l;
}

// ---------------- WMMA GEMM: C = A(MxK) @ B(KxN) [+bias] -------------------
// bf16x3 split (hi*hi + lo*hi + hi*lo), f32 accumulate. Inputs pre-split:
// A row-major bf16 hi/lo, B fragment-major packed bf16 hi/lo.
// Each wave32 computes a 16 x (16*NT) tile. K%32==0, M%16==0, N%(16*NT)==0.
template <int NT>
__global__ void wmma_gemm_kernel(const unsigned short* __restrict__ AH,
                                 const unsigned short* __restrict__ AL,
                                 const unsigned short* __restrict__ BH,
                                 const unsigned short* __restrict__ BL,
                                 const float* __restrict__ bias,
                                 float* __restrict__ C,
                                 int M, int N, int K) {
    int wave = (int)((blockIdx.x * blockDim.x + threadIdx.x) >> 5);
    int lane = (int)(threadIdx.x & 31);
    int tilesN = N / (16 * NT);
    int totalTiles = (M >> 4) * tilesN;
    if (wave >= totalTiles) return;               // wave-uniform: EXEC stays all-1s

    int tm   = (wave / tilesN) << 4;
    int tn   = (wave % tilesN) * (16 * NT);
    int half = lane >> 4;                          // 0: lanes 0-15, 1: lanes 16-31
    int l16  = lane & 15;
    int arow = tm + l16;                           // A-fragment row (ISA 7.12.2)

    v8f acc[NT];
    #pragma unroll
    for (int t = 0; t < NT; ++t) acc[t] = (v8f){};

    for (int kk = 0; kk < K; kk += 32) {
        size_t aoff = (size_t)arow * K + kk + half * 16;
        v16bf a_hi = ld_frag_a(AH + aoff);
        v16bf a_lo = ld_frag_a(AL + aoff);
        #pragma unroll
        for (int t = 0; t < NT; ++t) {
            int bcol = tn + t * 16 + l16;
            size_t boff = ((((size_t)(kk >> 5)) * N + bcol) * 2 + half) * 16;
            v16bf b_hi = ld_frag_b(BH + boff);
            v16bf b_lo = ld_frag_b(BL + boff);
            acc[t] = __builtin_amdgcn_wmma_f32_16x16x32_bf16(false, a_hi, false, b_hi,
                                                             (short)0, acc[t], false, false);
            acc[t] = __builtin_amdgcn_wmma_f32_16x16x32_bf16(false, a_lo, false, b_hi,
                                                             (short)0, acc[t], false, false);
            acc[t] = __builtin_amdgcn_wmma_f32_16x16x32_bf16(false, a_hi, false, b_lo,
                                                             (short)0, acc[t], false, false);
        }
    }
    // epilogue: C/D layout VGPR r -> row tm + r + half*8, col bcol
    #pragma unroll
    for (int t = 0; t < NT; ++t) {
        int bcol = tn + t * 16 + l16;
        float bv = bias ? bias[bcol] : 0.0f;       // hoisted: one load per column
        #pragma unroll
        for (int r = 0; r < 8; ++r) {
            int row = tm + r + half * 8;
            C[(size_t)row * N + bcol] = acc[t][r] + bv;
        }
    }
}

// ---------------- LayerNorm over 256 cols, one wave32 per row --------------
__global__ void ln_kernel(const float* __restrict__ x, const float* __restrict__ g,
                          const float* __restrict__ b, float* __restrict__ out,
                          float eps) {
    int row  = (int)blockIdx.x;
    int lane = (int)threadIdx.x;
    const float* xr = x + (size_t)row * DM;
    float v[8], s = 0.f;
    #pragma unroll
    for (int i = 0; i < 8; ++i) { v[i] = xr[lane + i * 32]; s += v[i]; }
    #pragma unroll
    for (int off = 16; off; off >>= 1) s += __shfl_xor(s, off, 32);
    float mu = s * (1.0f / DM);
    float q = 0.f;
    #pragma unroll
    for (int i = 0; i < 8; ++i) { float d = v[i] - mu; q += d * d; }
    #pragma unroll
    for (int off = 16; off; off >>= 1) q += __shfl_xor(q, off, 32);
    float rinv = rsqrtf(q * (1.0f / DM) + eps);
    #pragma unroll
    for (int i = 0; i < 8; ++i) {
        int c = lane + i * 32;
        out[(size_t)row * DM + c] = (v[i] - mu) * rinv * g[c] + b[c];
    }
}

// ---------------- causal depthwise conv (K=4) + SiLU -----------------------
__global__ void conv_silu_kernel(const float* __restrict__ xz, const float* __restrict__ w,
                                 const float* __restrict__ bias, float* __restrict__ xc) {
    int idx = (int)(blockIdx.x * blockDim.x + threadIdx.x);
    if (idx >= BB * LL * DI) return;
    int d = idx % DI;
    int l = (idx / DI) % LL;
    int b = idx / (DI * LL);
    float acc = bias[d];
    #pragma unroll
    for (int k = 0; k < DCONV; ++k) {
        int ls = l - (DCONV - 1) + k;
        if (ls >= 0)
            acc += xz[((size_t)(b * LL + ls)) * (2 * DI) + d] * w[d * DCONV + k];
    }
    float sg = 1.f / (1.f + __expf(-acc));
    xc[idx] = acc * sg;
}

// ---------------- dt = softplus(dt_raw @ W_dt + b_dt), K=16 ---------------
__global__ void dt_kernel(const float* __restrict__ xdbl, const float* __restrict__ Wdt,
                          const float* __restrict__ bdt, float* __restrict__ dt) {
    int idx = (int)(blockIdx.x * blockDim.x + threadIdx.x);
    if (idx >= MROWS * DI) return;
    int d = idx % DI;
    int m = idx / DI;
    const float* r = xdbl + (size_t)m * (DTR + 2 * DS);
    float acc = bdt[d];
    #pragma unroll
    for (int k = 0; k < DTR; ++k) acc += r[k] * Wdt[k * DI + d];
    dt[idx] = (acc > 20.f) ? acc : log1pf(__expf(acc));
}

// ---------------- selective scan: 1 thread per (b,d), 16 states in regs ----
#define SCAN_CHUNK 16
__global__ void scan_kernel(const float* __restrict__ xdbl, const float* __restrict__ dt,
                            const float* __restrict__ xc, const float* __restrict__ xz,
                            const float* __restrict__ A_log, const float* __restrict__ Dp,
                            float* __restrict__ y) {
    int b = (int)blockIdx.x;      // 8 blocks
    int d = (int)threadIdx.x;     // 512 threads
    __shared__ float sBC[SCAN_CHUNK * 32];   // per-step: Bm[0..15], Cm[0..15]

    float A[DS], h[DS];
    #pragma unroll
    for (int s = 0; s < DS; ++s) { A[s] = -__expf(A_log[d * DS + s]); h[s] = 0.f; }
    float Dval = Dp[d];

    for (int l0 = 0; l0 < LL; l0 += SCAN_CHUNK) {
        __syncthreads();
        if (threadIdx.x < SCAN_CHUNK * 32) {
            int rr = (int)(threadIdx.x >> 5), cc = (int)(threadIdx.x & 31);
            sBC[rr * 32 + cc] =
                xdbl[((size_t)(b * LL + l0 + rr)) * (DTR + 2 * DS) + DTR + cc];
        }
        __syncthreads();
        if (l0 + SCAN_CHUNK < LL)   // hide latency of next chunk (global_prefetch)
            __builtin_prefetch(&dt[((size_t)(b * LL + l0 + SCAN_CHUNK)) * DI + d], 0, 1);

        for (int li = 0; li < SCAN_CHUNK; ++li) {
            size_t mi = (size_t)(b * LL + l0 + li);
            float dtv = dt[mi * DI + d];
            float xv  = xc[mi * DI + d];
            float zv  = xz[mi * (2 * DI) + DI + d];
            float acc = 0.f;
            #pragma unroll
            for (int s = 0; s < DS; ++s) {
                float dA = __expf(dtv * A[s]);
                h[s] = dA * h[s] + dtv * sBC[li * 32 + s] * xv;
                acc += h[s] * sBC[li * 32 + 16 + s];
            }
            float yv = acc + xv * Dval;
            float sg = 1.f / (1.f + __expf(-zv));
            y[mi * DI + d] = yv * (zv * sg);
        }
    }
}

// ---------------- ReLU(gemm_out) + residual --------------------------------
__global__ void relu_res_kernel(const float* __restrict__ go, const float* __restrict__ x,
                                float* __restrict__ a, int n) {
    int i = (int)(blockIdx.x * blockDim.x + threadIdx.x);
    if (i < n) a[i] = fmaxf(go[i], 0.f) + x[i];
}

// ---------------- fusion gate pieces ---------------------------------------
__global__ void pool_kernel(const float* __restrict__ a, const float* __restrict__ e,
                            float* __restrict__ pooled) {
    int idx = (int)(blockIdx.x * blockDim.x + threadIdx.x);   // BB*DM
    if (idx >= BB * DM) return;
    int c = idx % DM, b = idx / DM;
    float s = 0.f;
    for (int l = 0; l < LL; ++l) {
        size_t mi = (size_t)(b * LL + l) * DM + c;
        s += a[mi] + e[mi];
    }
    pooled[idx] = s * (0.5f / LL);
}

__global__ void gate_kernel(const float* __restrict__ pooled, const float* __restrict__ W,
                            float* __restrict__ gate) {
    int idx = (int)(blockIdx.x * blockDim.x + threadIdx.x);   // BB*DM
    if (idx >= BB * DM) return;
    int n = idx % DM, b = idx / DM;
    float acc = 0.f;
    for (int c = 0; c < DM; ++c) acc += pooled[b * DM + c] * W[c * DM + n];
    gate[idx] = 1.f / (1.f + __expf(-acc));
}

__global__ void final_kernel(const float* __restrict__ a, const float* __restrict__ ident,
                             const float* __restrict__ gate, float* __restrict__ out) {
    int idx = (int)(blockIdx.x * blockDim.x + threadIdx.x);   // MROWS*DM
    if (idx >= MROWS * DM) return;
    int c = idx % DM;
    int b = (idx / DM) / LL;
    out[idx] = a[idx] * gate[b * DM + c] + ident[idx];
}

// ---------------- host orchestration ---------------------------------------
static inline int cdiv(int a, int b) { return (a + b - 1) / b; }

static void launch_gemm(const unsigned short* AH, const unsigned short* AL,
                        const unsigned short* BH, const unsigned short* BL,
                        const float* bias, float* C, int M, int N, int K, hipStream_t s) {
    if (N % 64 == 0) {
        int tiles = (M >> 4) * (N / 64);
        wmma_gemm_kernel<4><<<cdiv(tiles, 8), 256, 0, s>>>(AH, AL, BH, BL, bias, C, M, N, K);
    } else if (N % 48 == 0) {
        int tiles = (M >> 4) * (N / 48);
        wmma_gemm_kernel<3><<<cdiv(tiles, 8), 256, 0, s>>>(AH, AL, BH, BL, bias, C, M, N, K);
    } else {
        int tiles = (M >> 4) * (N >> 4);
        wmma_gemm_kernel<1><<<cdiv(tiles, 8), 256, 0, s>>>(AH, AL, BH, BL, bias, C, M, N, K);
    }
}

extern "C" void kernel_launch(void* const* d_in, const int* in_sizes, int n_in,
                              void* d_out, int out_size, void* d_ws, size_t ws_size,
                              hipStream_t stream) {
    (void)in_sizes; (void)n_in; (void)out_size; (void)ws_size;

    const float* x_in[2]  = { (const float*)d_in[0], (const float*)d_in[1] };
    const float* sc_W     = (const float*)d_in[24];
    const float* sc_b     = (const float*)d_in[25];
    const float* sc_ln_g  = (const float*)d_in[26];
    const float* sc_ln_b  = (const float*)d_in[27];
    const float* fusion_W = (const float*)d_in[28];

    // workspace layout; intermediates shared across branches
    float* ws = (float*)d_ws;
    size_t off = 0;
    float* XN   = ws + off; off += (size_t)MROWS * DM;          // LN(x)
    float* XZ   = ws + off; off += (size_t)MROWS * 2 * DI;      // x @ W_in
    float* XC   = ws + off; off += (size_t)MROWS * DI;          // conv+silu
    float* XDBL = ws + off; off += (size_t)MROWS * (DTR + 2*DS);// xc @ W_x
    float* DT   = ws + off; off += (size_t)MROWS * DI;          // softplus dt
    float* Y    = ws + off; off += (size_t)MROWS * DI;          // scan output
    float* GO   = ws + off; off += (size_t)MROWS * DM;          // y @ W_out
    float* SCT  = ws + off; off += (size_t)MROWS * DM;          // shortcut pre-LN
    float* ABUF[2]; ABUF[0] = ws + off; off += (size_t)MROWS * DM;
                    ABUF[1] = ws + off; off += (size_t)MROWS * DM;
    float* IDB[2];  IDB[0]  = ws + off; off += (size_t)MROWS * DM;
                    IDB[1]  = ws + off; off += (size_t)MROWS * DM;
    float* POOL = ws + off; off += (size_t)BB * DM;
    float* GATE = ws + off; off += (size_t)BB * DM;
    // bf16 hi/lo staging (2 ushorts per float slot)
    unsigned short* AH  = (unsigned short*)(ws + off); off += (size_t)MROWS * DI / 2;
    unsigned short* AL  = (unsigned short*)(ws + off); off += (size_t)MROWS * DI / 2;
    unsigned short* BH  = (unsigned short*)(ws + off); off += (size_t)DM * 2 * DI / 2;
    unsigned short* BL  = (unsigned short*)(ws + off); off += (size_t)DM * 2 * DI / 2;
    unsigned short* SWH = (unsigned short*)(ws + off); off += (size_t)DM * DM / 2;
    unsigned short* SWL = (unsigned short*)(ws + off); off += (size_t)DM * DM / 2;

    const int ELT = 256;
    // shared shortcut weight: pack once
    pack_b_kernel<<<cdiv(DM * DM, ELT), ELT, 0, stream>>>(sc_W, SWH, SWL, DM, DM);

    for (int br = 0; br < 2; ++br) {
        int base = 2 + br * 11;
        const float* ln_g   = (const float*)d_in[base + 0];
        const float* ln_b   = (const float*)d_in[base + 1];
        const float* W_in   = (const float*)d_in[base + 2];
        const float* conv_w = (const float*)d_in[base + 3];
        const float* conv_b = (const float*)d_in[base + 4];
        const float* W_x    = (const float*)d_in[base + 5];
        const float* W_dt   = (const float*)d_in[base + 6];
        const float* b_dt   = (const float*)d_in[base + 7];
        const float* A_log  = (const float*)d_in[base + 8];
        const float* Dp     = (const float*)d_in[base + 9];
        const float* W_out  = (const float*)d_in[base + 10];
        const float* x      = x_in[br];

        // shared shortcut: LN(x @ sc_W + sc_b), eps=1e-5
        split_a_kernel<<<cdiv(MROWS * DM, ELT), ELT, 0, stream>>>(x, AH, AL, MROWS * DM);
        launch_gemm(AH, AL, SWH, SWL, sc_b, SCT, MROWS, DM, DM, stream);
        ln_kernel<<<MROWS, 32, 0, stream>>>(SCT, sc_ln_g, sc_ln_b, IDB[br], 1e-5f);

        // branch body
        ln_kernel<<<MROWS, 32, 0, stream>>>(x, ln_g, ln_b, XN, 1e-6f);
        split_a_kernel<<<cdiv(MROWS * DM, ELT), ELT, 0, stream>>>(XN, AH, AL, MROWS * DM);
        pack_b_kernel<<<cdiv(DM * 2 * DI, ELT), ELT, 0, stream>>>(W_in, BH, BL, 2 * DI, DM);
        launch_gemm(AH, AL, BH, BL, nullptr, XZ, MROWS, 2 * DI, DM, stream);

        conv_silu_kernel<<<cdiv(BB * LL * DI, ELT), ELT, 0, stream>>>(XZ, conv_w, conv_b, XC);

        split_a_kernel<<<cdiv(MROWS * DI, ELT), ELT, 0, stream>>>(XC, AH, AL, MROWS * DI);
        pack_b_kernel<<<cdiv(DI * (DTR + 2 * DS), ELT), ELT, 0, stream>>>(W_x, BH, BL,
                                                                          DTR + 2 * DS, DI);
        launch_gemm(AH, AL, BH, BL, nullptr, XDBL, MROWS, DTR + 2 * DS, DI, stream);

        dt_kernel<<<cdiv(MROWS * DI, ELT), ELT, 0, stream>>>(XDBL, W_dt, b_dt, DT);
        scan_kernel<<<BB, DI, 0, stream>>>(XDBL, DT, XC, XZ, A_log, Dp, Y);

        split_a_kernel<<<cdiv(MROWS * DI, ELT), ELT, 0, stream>>>(Y, AH, AL, MROWS * DI);
        pack_b_kernel<<<cdiv(DI * DM, ELT), ELT, 0, stream>>>(W_out, BH, BL, DM, DI);
        launch_gemm(AH, AL, BH, BL, nullptr, GO, MROWS, DM, DI, stream);

        relu_res_kernel<<<cdiv(MROWS * DM, ELT), ELT, 0, stream>>>(GO, x, ABUF[br], MROWS * DM);
    }

    // fusion gate + residual add, write both outputs
    pool_kernel<<<cdiv(BB * DM, ELT), ELT, 0, stream>>>(ABUF[0], ABUF[1], POOL);
    gate_kernel<<<cdiv(BB * DM, ELT), ELT, 0, stream>>>(POOL, fusion_W, GATE);
    float* out = (float*)d_out;
    final_kernel<<<cdiv(MROWS * DM, ELT), ELT, 0, stream>>>(ABUF[0], IDB[0], GATE, out);
    final_kernel<<<cdiv(MROWS * DM, ELT), ELT, 0, stream>>>(ABUF[1], IDB[1], GATE,
                                                            out + (size_t)MROWS * DM);
}
